// NeuralMemory_15152644620289
// MI455X (gfx1250) — compile-verified, matrix operand order checked by reading, not run on hardware
//
#include <hip/hip_runtime.h>
#include <hip/hip_bf16.h>
#include <math.h>

// ---------------------------------------------------------------------------
// Types for CDNA5 WMMA (wave32, 16x16x32 bf16 -> f32)
// ---------------------------------------------------------------------------
typedef __bf16 bf16;
typedef __attribute__((ext_vector_type(16))) __bf16 v16bf;
typedef __attribute__((ext_vector_type(8)))  __bf16 v8bf;
typedef __attribute__((ext_vector_type(8)))  float   v8f;
typedef unsigned int v4u __attribute__((ext_vector_type(4)));
typedef int          v8i __attribute__((ext_vector_type(8)));
typedef int          v4i __attribute__((ext_vector_type(4)));

#define WMMA_BF16(a, b, c) \
  __builtin_amdgcn_wmma_f32_16x16x32_bf16(false, (a), false, (b), (short)0, (c), false, false)

#if defined(__has_builtin)
#if __has_builtin(__builtin_amdgcn_tensor_load_to_lds) && \
    __has_builtin(__builtin_amdgcn_s_wait_tensorcnt)
#define USE_TDM 1
#endif
#endif
#ifndef USE_TDM
#define USE_TDM 0
#endif

__device__ __forceinline__ float sigmoidf_(float x) { return 1.f / (1.f + __expf(-x)); }

// ---------------------------------------------------------------------------
// TDM: issue a 2D tile load (tileD1 rows x tileD0 cols, bf16) into LDS.
// D# per CDNA5 ISA. Padding: every 16 DWORDs of
// tile row (=32 bf16) insert 8 DWORDs -> LDS row stride 48 halves (96B),
// matching the bank-spread fragment layout the compute reads use.
// This toolchain exposes the 6-arg builtin:
//   (uint32x4 g0, int32x8 g1, int32x4 g2, int32x4 g3, int32x8 pad, i32 cpol)
// ---------------------------------------------------------------------------
#if USE_TDM
__device__ __forceinline__ void tdm_load_2d(unsigned ldsOff, const bf16* gaddr,
                                            unsigned rowStrideElems,
                                            unsigned tileD0, unsigned tileD1) {
  unsigned long long ga = (unsigned long long)(uintptr_t)gaddr;
  v4u g0;
  g0[0] = 1u;                                   // count=1, user desc, no gather
  g0[1] = ldsOff;                               // lds_addr (bytes)
  g0[2] = (unsigned)(ga & 0xFFFFFFFFu);         // global_addr[31:0]
  g0[3] = (unsigned)((ga >> 32) & 0x01FFFFFFu)  // global_addr[56:32]
          | (2u << 30);                         // type = 2 ("image")
  v8i g1;
  g1[0] = (int)((1u << 16)                      // data_size = 1 -> 2 bytes
                | (1u << 20)                    // pad_enable
                | (3u << 22)                    // pad_interval: 16 DWORDs
                | (7u << 25));                  // pad_amount: 8 DWORDs
  g1[1] = (int)((tileD0 & 0xFFFFu) << 16);      // tensor_dim0 = tile width
  g1[2] = (int)((tileD0 >> 16) | ((tileD1 & 0xFFFFu) << 16));  // tensor_dim1
  g1[3] = (int)((tileD1 >> 16) | ((tileD0 & 0xFFFFu) << 16));  // tile_dim0
  g1[4] = (int)(tileD1 & 0xFFFFu);              // tile_dim1 (tile_dim2 = 0)
  g1[5] = (int)rowStrideElems;                  // tensor_dim0_stride low32
  g1[6] = 0;
  g1[7] = 0;
  v4i g2 = {0, 0, 0, 0};
  v4i g3 = {0, 0, 0, 0};
  v8i g4 = {0, 0, 0, 0, 0, 0, 0, 0};
  __builtin_amdgcn_tensor_load_to_lds(g0, g1, g2, g3, g4, 0);
}
#endif

// ---------------------------------------------------------------------------
// GEMM (NT): C[M,N] = A[M,K] * Bp[N,K]^T  (+bias, +resid, optional SiLU)
//   A, Bp bf16 (K-contiguous).  64(M) x 128(N) tile per 256-thread block.
//   TDM path: wave 0 drives a double-buffered tensor DMA pipeline into
//   dynamic LDS; all waves consume via WMMA. Fallback: synchronous staging.
//   act==0: outF = val (f32), outB = bf16(val)
//   act==1: outB = bf16(silu(val)), outB2 = bf16(val)   (pre-activation z)
// ---------------------------------------------------------------------------
// Dynamic LDS layout (row stride 48 halves = 96B):
//   A buffers: 2 x 64*96  = 2 x 6144 B  at offsets 0, 6144
//   B buffers: 2 x 128*96 = 2 x 12288 B at offsets 12288, 24576
#define GEMM_NT_LDS_BYTES 36864

__global__ __launch_bounds__(256) void gemm_nt_kernel(
    const bf16* __restrict__ A, const bf16* __restrict__ Bp,
    const float* __restrict__ bias, const float* __restrict__ resid,
    float* __restrict__ outF, bf16* __restrict__ outB, bf16* __restrict__ outB2,
    int M, int N, int K, int act)
{
  extern __shared__ char smem[];

  const int tid   = threadIdx.x;
  const int lane  = tid & 31;
  const int wave  = tid >> 5;
  const int waveM = wave & 1;   // 2 waves along M (32 rows each)
  const int waveN = wave >> 1;  // 4 waves along N (32 cols each)
  const int hh    = lane >> 4;  // K-half selector
  const int l16   = lane & 15;
  const size_t tileM = (size_t)blockIdx.y * 64;
  const size_t tileN = (size_t)blockIdx.x * 128;

  v8f acc[2][2] = {};
  const int nSteps = K >> 5;

#if USE_TDM
  const unsigned ldsBase = __builtin_amdgcn_groupstaticsize();
  // Prologue: stage 0 into buffer 0
  if (wave == 0) {
    tdm_load_2d(ldsBase + 0,     A  + tileM * (size_t)K, (unsigned)K, 32u, 64u);
    tdm_load_2d(ldsBase + 12288, Bp + tileN * (size_t)K, (unsigned)K, 32u, 128u);
  }
  for (int s = 0; s < nSteps; ++s) {
    const int ib = s & 1;
    if (wave == 0) {
      if (s + 1 < nSteps) {
        const int k1 = (s + 1) << 5;
        tdm_load_2d(ldsBase + (ib ? 0u : 6144u),
                    A + tileM * (size_t)K + k1, (unsigned)K, 32u, 64u);
        tdm_load_2d(ldsBase + (ib ? 12288u : 24576u),
                    Bp + tileN * (size_t)K + k1, (unsigned)K, 32u, 128u);
        __builtin_amdgcn_s_wait_tensorcnt(2);   // stage s complete
      } else {
        __builtin_amdgcn_s_wait_tensorcnt(0);
      }
    }
    __syncthreads();
    const bf16* As = (const bf16*)(smem + (ib ? 6144 : 0));
    const bf16* Bs = (const bf16*)(smem + 12288 + (ib ? 12288 : 0));

    v16bf a0 = *(const v16bf*)(As + (size_t)(waveM * 32 + 0  + l16) * 48 + hh * 16);
    v16bf a1 = *(const v16bf*)(As + (size_t)(waveM * 32 + 16 + l16) * 48 + hh * 16);
    v16bf b0 = *(const v16bf*)(Bs + (size_t)(waveN * 32 + 0  + l16) * 48 + hh * 16);
    v16bf b1 = *(const v16bf*)(Bs + (size_t)(waveN * 32 + 16 + l16) * 48 + hh * 16);

    acc[0][0] = WMMA_BF16(a0, b0, acc[0][0]);
    acc[0][1] = WMMA_BF16(a0, b1, acc[0][1]);
    acc[1][0] = WMMA_BF16(a1, b0, acc[1][0]);
    acc[1][1] = WMMA_BF16(a1, b1, acc[1][1]);
    __syncthreads();   // all reads of buf[ib] done before it is re-filled
  }
#else
  bf16* As = (bf16*)smem;              // [64][48]
  bf16* Bs = (bf16*)(smem + 12288);    // [128][48]
  const int arow = tid >> 2, acol = (tid & 3) << 3;
  for (int s = 0; s < nSteps; ++s) {
    const int k0 = s << 5;
    *(v8bf*)(As + (size_t)arow * 48 + acol) =
        *(const v8bf*)(A + (tileM + arow) * (size_t)K + k0 + acol);
#pragma unroll
    for (int c = 0; c < 2; ++c) {
      int ch = tid + (c << 8);
      int br = ch >> 2, bc = (ch & 3) << 3;
      *(v8bf*)(Bs + (size_t)br * 48 + bc) =
          *(const v8bf*)(Bp + (tileN + br) * (size_t)K + k0 + bc);
    }
    if (k0 + 32 < K)
      __builtin_prefetch(A + (tileM + arow) * (size_t)K + k0 + 32 + acol, 0, 1);
    __syncthreads();
    v16bf a0 = *(const v16bf*)(As + (size_t)(waveM * 32 + 0  + l16) * 48 + hh * 16);
    v16bf a1 = *(const v16bf*)(As + (size_t)(waveM * 32 + 16 + l16) * 48 + hh * 16);
    v16bf b0 = *(const v16bf*)(Bs + (size_t)(waveN * 32 + 0  + l16) * 48 + hh * 16);
    v16bf b1 = *(const v16bf*)(Bs + (size_t)(waveN * 32 + 16 + l16) * 48 + hh * 16);
    acc[0][0] = WMMA_BF16(a0, b0, acc[0][0]);
    acc[0][1] = WMMA_BF16(a0, b1, acc[0][1]);
    acc[1][0] = WMMA_BF16(a1, b0, acc[1][0]);
    acc[1][1] = WMMA_BF16(a1, b1, acc[1][1]);
    __syncthreads();
  }
#endif

#pragma unroll
  for (int i = 0; i < 2; ++i)
#pragma unroll
    for (int j = 0; j < 2; ++j) {
      const size_t gn = tileN + waveN * 32 + j * 16 + l16;
      const float bv = bias ? bias[gn] : 0.f;
#pragma unroll
      for (int r = 0; r < 8; ++r) {
        const size_t gm  = tileM + waveM * 32 + i * 16 + hh * 8 + r;
        const size_t idx = gm * (size_t)N + gn;
        float val = acc[i][j][r] + bv;
        if (resid) val += resid[idx];
        if (act == 1) {
          float s = sigmoidf_(val);
          if (outB)  outB[idx]  = (bf16)(val * s);
          if (outB2) outB2[idx] = (bf16)val;
        } else {
          if (outF) outF[idx] = val;
          if (outB) outB[idx] = (bf16)val;
        }
      }
    }
}

// ---------------------------------------------------------------------------
// GEMM (TN): C[I,J] = sum_m A[m,i] * B[m,j]   (dW = dY^T @ X, K = M = 8192)
//   LDS-transposed staging so WMMA fragments read contiguously.
// ---------------------------------------------------------------------------
__global__ __launch_bounds__(256) void gemm_tn_kernel(
    const bf16* __restrict__ A, const bf16* __restrict__ B,
    float* __restrict__ Cmat, int M, int I, int J)
{
  __shared__ alignas(32) bf16 At[64][48];
  __shared__ alignas(32) bf16 Bt[128][48];

  const int tid   = threadIdx.x;
  const int lane  = tid & 31;
  const int wave  = tid >> 5;
  const int waveI = wave & 1;
  const int waveJ = wave >> 1;
  const int hh    = lane >> 4;
  const int l16   = lane & 15;
  const size_t tileI = (size_t)blockIdx.y * 64;
  const size_t tileJ = (size_t)blockIdx.x * 128;

  const int arow = tid >> 3, aic = (tid & 7) << 3;  // 32 m-rows x 8 chunks(8)

  v8f acc[2][2] = {};

  for (int m0 = 0; m0 < M; m0 += 32) {
    v8bf va = *(const v8bf*)(A + (size_t)(m0 + arow) * I + tileI + aic);
#pragma unroll
    for (int e = 0; e < 8; ++e) At[aic + e][arow] = va[e];
#pragma unroll
    for (int c = 0; c < 2; ++c) {
      int ch = tid + (c << 8);
      int brow = ch >> 4, bjc = (ch & 15) << 3;
      v8bf vb = *(const v8bf*)(B + (size_t)(m0 + brow) * J + tileJ + bjc);
#pragma unroll
      for (int e = 0; e < 8; ++e) Bt[bjc + e][brow] = vb[e];
    }
    if (m0 + 32 < M) {
      __builtin_prefetch(A + (size_t)(m0 + 32 + arow) * I + tileI + aic, 0, 1);
      __builtin_prefetch(B + (size_t)(m0 + 32 + arow) * J + tileJ, 0, 1);
    }
    __syncthreads();

    v16bf a0 = *(const v16bf*)&At[waveI * 32 + 0  + l16][hh * 16];
    v16bf a1 = *(const v16bf*)&At[waveI * 32 + 16 + l16][hh * 16];
    v16bf b0 = *(const v16bf*)&Bt[waveJ * 32 + 0  + l16][hh * 16];
    v16bf b1 = *(const v16bf*)&Bt[waveJ * 32 + 16 + l16][hh * 16];

    acc[0][0] = WMMA_BF16(a0, b0, acc[0][0]);
    acc[0][1] = WMMA_BF16(a0, b1, acc[0][1]);
    acc[1][0] = WMMA_BF16(a1, b0, acc[1][0]);
    acc[1][1] = WMMA_BF16(a1, b1, acc[1][1]);
    __syncthreads();
  }

#pragma unroll
  for (int i = 0; i < 2; ++i)
#pragma unroll
    for (int j = 0; j < 2; ++j) {
      const size_t gj = tileJ + waveJ * 32 + j * 16 + l16;
#pragma unroll
      for (int r = 0; r < 8; ++r) {
        const size_t gi = tileI + waveI * 32 + i * 16 + hh * 8 + r;
        Cmat[gi * (size_t)J + gj] = acc[i][j][r];
      }
    }
}

// ---------------------------------------------------------------------------
// Small kernels
// ---------------------------------------------------------------------------
__global__ void zero_kernel(float* p, int n) {
  if ((int)threadIdx.x < n) p[threadIdx.x] = 0.f;
}

// f32 -> bf16 pack (optional transpose): dst is [C,R] when transpose, else [R,C]
__global__ void pack_kernel(const float* __restrict__ src, bf16* __restrict__ dst,
                            int R, int C, int transpose) {
  size_t i = (size_t)blockIdx.x * 256 + threadIdx.x;
  if (i >= (size_t)R * C) return;
  int r = (int)(i / C), c = (int)(i % C);
  bf16 v = (bf16)src[i];
  if (transpose) dst[(size_t)c * R + r] = v;
  else           dst[i] = v;
}

// Per-row L2 normalize in place, also emit bf16 copy. One wave per row.
__global__ void l2norm_kernel(float* __restrict__ x, bf16* __restrict__ xb, int H) {
  int row  = blockIdx.x * 8 + (threadIdx.x >> 5);
  int lane = threadIdx.x & 31;
  float* p = x + (size_t)row * H;
  float ss = 0.f;
  for (int c = lane; c < H; c += 32) { float v = p[c]; ss += v * v; }
  for (int o = 16; o > 0; o >>= 1) ss += __shfl_xor(ss, o, 32);
  float inv = 1.f / fmaxf(sqrtf(ss), 1e-12f);
  for (int c = lane; c < H; c += 32) {
    float v = p[c] * inv;
    p[c] = v;
    xb[(size_t)row * H + c] = (bf16)v;
  }
}

// Gate projections: lr = 0.1*sigmoid(x.wlr+blr); accumulate global sums of
// lr, forget, mom into sums[0..2]. One wave per token.
__global__ void gates_kernel(const float* __restrict__ x,
                             const float* __restrict__ wlr, const float* __restrict__ blr,
                             const float* __restrict__ wf,  const float* __restrict__ bfp,
                             const float* __restrict__ wm,  const float* __restrict__ bm,
                             float* __restrict__ lr, float* __restrict__ sums, int H) {
  int m    = blockIdx.x * 8 + (threadIdx.x >> 5);
  int lane = threadIdx.x & 31;
  const float* xr = x + (size_t)m * H;
  float s0 = 0.f, s1 = 0.f, s2 = 0.f;
  for (int c = lane; c < H; c += 32) {
    float v = xr[c];
    s0 += v * wlr[c];
    s1 += v * wf[c];
    s2 += v * wm[c];
  }
  for (int o = 16; o > 0; o >>= 1) {
    s0 += __shfl_xor(s0, o, 32);
    s1 += __shfl_xor(s1, o, 32);
    s2 += __shfl_xor(s2, o, 32);
  }
  if (lane == 0) {
    float l = 0.1f * sigmoidf_(s0 + blr[0]);
    float f = sigmoidf_(s1 + bfp[0]);
    float g = sigmoidf_(s2 + bm[0]);
    lr[m] = l;
    atomicAdd(&sums[0], l);
    atomicAdd(&sums[1], f);
    atomicAdd(&sums[2], g);
  }
}

// dpred = scale * lr[token] * (pred - v); scale = 2/(H*C) folds loss + chunk mean
__global__ void dpred_kernel(const float* __restrict__ pred, const float* __restrict__ v,
                             const float* __restrict__ lr, float* __restrict__ dpF,
                             bf16* __restrict__ dpB, int H, float scale, size_t n) {
  size_t i = (size_t)blockIdx.x * 256 + threadIdx.x;
  if (i >= n) return;
  size_t m = i / (size_t)H;
  float d = scale * lr[m] * (pred[i] - v[i]);
  dpF[i] = d;
  dpB[i] = (bf16)d;
}

// dz = dh * silu'(z)
__global__ void dsilu_kernel(const float* __restrict__ dh, const bf16* __restrict__ zb,
                             bf16* __restrict__ dzb, size_t n) {
  size_t i = (size_t)blockIdx.x * 256 + threadIdx.x;
  if (i >= n) return;
  float z = (float)zb[i];
  float s = sigmoidf_(z);
  float g = s * (1.f + z * (1.f - s));
  dzb[i] = (bf16)(dh[i] * g);
}

// Column sum (bias gradients)
template <typename T>
__global__ void colsum_kernel(const T* __restrict__ src, float* __restrict__ db,
                              int rows, int N) {
  __shared__ float red[256];
  int col = blockIdx.x;
  float s = 0.f;
  for (int r = threadIdx.x; r < rows; r += 256) s += (float)src[(size_t)r * N + col];
  red[threadIdx.x] = s;
  __syncthreads();
  for (int o = 128; o > 0; o >>= 1) {
    if ((int)threadIdx.x < o) red[threadIdx.x] += red[threadIdx.x + o];
    __syncthreads();
  }
  if (threadIdx.x == 0) db[col] = red[0];
}

// new_p = p*(1 - mean(forget)) + mean(mom)*m - mean(lr)*g
__global__ void update_kernel(const float* __restrict__ p, const float* __restrict__ g,
                              const float* __restrict__ mom, const float* __restrict__ sums,
                              float invM, float* __restrict__ outF, bf16* __restrict__ outB,
                              size_t n) {
  size_t i = (size_t)blockIdx.x * 256 + threadIdx.x;
  if (i >= n) return;
  float tbar = sums[0] * invM;  // mean lr (already *0.1)
  float abar = sums[1] * invM;  // mean forget
  float ebar = sums[2] * invM;  // mean momentum gate
  float np = p[i] * (1.f - abar) + ebar * mom[i] - tbar * g[i];
  if (outF) outF[i] = np;
  if (outB) outB[i] = (bf16)np;
}

// ---------------------------------------------------------------------------
// Host orchestration
// ---------------------------------------------------------------------------
extern "C" void kernel_launch(void* const* d_in, const int* in_sizes, int n_in,
                              void* d_out, int out_size, void* d_ws, size_t ws_size,
                              hipStream_t stream) {
  (void)in_sizes; (void)n_in; (void)out_size; (void)ws_size;
  const float* x      = (const float*)d_in[0];
  const float* wq     = (const float*)d_in[1];
  const float* bq     = (const float*)d_in[2];
  const float* wk     = (const float*)d_in[3];
  const float* bk     = (const float*)d_in[4];
  const float* wv     = (const float*)d_in[5];
  const float* bv     = (const float*)d_in[6];
  const float* wlr    = (const float*)d_in[7];
  const float* blr    = (const float*)d_in[8];
  const float* wf     = (const float*)d_in[9];
  const float* bfp    = (const float*)d_in[10];
  const float* wm     = (const float*)d_in[11];
  const float* bm     = (const float*)d_in[12];
  const float* mw1    = (const float*)d_in[13];
  const float* mb1    = (const float*)d_in[14];
  const float* mw2    = (const float*)d_in[15];
  const float* mb2    = (const float*)d_in[16];
  const float* mom_w1 = (const float*)d_in[17];
  const float* mom_b1 = (const float*)d_in[18];
  const float* mom_w2 = (const float*)d_in[19];
  const float* mom_b2 = (const float*)d_in[20];
  float* out = (float*)d_out;

  const int Bb = 4, S = 2048, H = 512, HE = 2048, Cc = 8;
  const int M = Bb * S;                  // 8192 tokens
  const size_t MH   = (size_t)M * H;
  const size_t MHE  = (size_t)M * HE;
  const size_t W1SZ = (size_t)HE * H;    // per-depth weight size
  const size_t HH   = (size_t)H * H;

  // Workspace bump allocator (256B aligned)
  char* base = (char*)d_ws;
  size_t off = 0;
  auto alloc = [&](size_t bytes) -> void* {
    void* p = base + off;
    off = (off + bytes + 255) & ~(size_t)255;
    return p;
  };

  bf16* xb    = (bf16*)alloc(MH * 2);
  bf16* qb    = (bf16*)alloc(MH * 2);
  bf16* kb    = (bf16*)alloc(MH * 2);
  bf16* x1b   = (bf16*)alloc(MH * 2);
  bf16* dpb   = (bf16*)alloc(MH * 2);     // dpred bf16; reused as x1q bf16
  bf16* dx1b  = (bf16*)alloc(MH * 2);
  bf16* h0b   = (bf16*)alloc(MHE * 2);    // reused in final forward
  bf16* h1b   = (bf16*)alloc(MHE * 2);    // reused in final forward
  bf16* z0b   = (bf16*)alloc(MHE * 2);
  bf16* z1b   = (bf16*)alloc(MHE * 2);
  bf16* dz0b  = (bf16*)alloc(MHE * 2);
  bf16* dz1b  = (bf16*)alloc(MHE * 2);
  bf16* wqp   = (bf16*)alloc(HH * 2);
  bf16* wkp   = (bf16*)alloc(HH * 2);
  bf16* wvp   = (bf16*)alloc(HH * 2);
  bf16* mw1p  = (bf16*)alloc(2 * W1SZ * 2);
  bf16* mw2p  = (bf16*)alloc(2 * W1SZ * 2);
  bf16* mw1T  = (bf16*)alloc(2 * W1SZ * 2);
  bf16* mw2T  = (bf16*)alloc(2 * W1SZ * 2);
  bf16* nmw1p = (bf16*)alloc(2 * W1SZ * 2);
  bf16* nmw2p = (bf16*)alloc(2 * W1SZ * 2);
  float* qf     = (float*)alloc(MH * 4);
  float* kf     = (float*)alloc(MH * 4);
  float* vf     = (float*)alloc(MH * 4);
  float* x1f    = (float*)alloc(MH * 4);
  float* predf  = (float*)alloc(MH * 4);
  float* dpredf = (float*)alloc(MH * 4);  // reused as x1q f32
  float* dhf    = (float*)alloc(MHE * 4);
  float* lrv    = (float*)alloc((size_t)M * 4);
  float* gw1    = (float*)alloc(2 * W1SZ * 4);
  float* gw2    = (float*)alloc(2 * W1SZ * 4);
  float* gb1    = (float*)alloc(2 * (size_t)HE * 4);
  float* gb2    = (float*)alloc(2 * (size_t)H * 4);
  float* nmb1   = (float*)alloc(2 * (size_t)HE * 4);
  float* nmb2   = (float*)alloc(2 * (size_t)H * 4);
  float* sums   = (float*)alloc(4 * 4);

  const dim3 blk(256);
  auto gemmNT = [&](const bf16* A, const bf16* Bw, const float* bias, const float* resid,
                    float* oF, bf16* oB, bf16* oB2, int Mm, int Nn, int Kk, int act) {
    gemm_nt_kernel<<<dim3(Nn / 128, Mm / 64), blk, GEMM_NT_LDS_BYTES, stream>>>(
        A, Bw, bias, resid, oF, oB, oB2, Mm, Nn, Kk, act);
  };
  auto gemmTN = [&](const bf16* A, const bf16* Bw, float* Cw, int Mm, int Ii, int Jj) {
    gemm_tn_kernel<<<dim3(Jj / 128, Ii / 64), blk, 0, stream>>>(A, Bw, Cw, Mm, Ii, Jj);
  };
  auto ew = [&](size_t n) { return dim3((unsigned)((n + 255) / 256)); };

  // --- Phase 0: init + packs -------------------------------------------------
  zero_kernel<<<1, 32, 0, stream>>>(sums, 4);
  pack_kernel<<<ew(MH), blk, 0, stream>>>(x, xb, M, H, 0);
  pack_kernel<<<ew(HH), blk, 0, stream>>>(wq, wqp, H, H, 0);
  pack_kernel<<<ew(HH), blk, 0, stream>>>(wk, wkp, H, H, 0);
  pack_kernel<<<ew(HH), blk, 0, stream>>>(wv, wvp, H, H, 0);
  pack_kernel<<<ew(2 * W1SZ), blk, 0, stream>>>(mw1, mw1p, 2 * HE, H, 0);
  pack_kernel<<<ew(2 * W1SZ), blk, 0, stream>>>(mw2, mw2p, 2 * H, HE, 0);
  for (int d = 0; d < 2; ++d) {
    pack_kernel<<<ew(W1SZ), blk, 0, stream>>>(mw1 + d * W1SZ, mw1T + d * W1SZ, HE, H, 1);
    pack_kernel<<<ew(W1SZ), blk, 0, stream>>>(mw2 + d * W1SZ, mw2T + d * W1SZ, H, HE, 1);
  }

  // --- Phase 1: projections + gates -----------------------------------------
  gemmNT(xb, wqp, bq, nullptr, qf, nullptr, nullptr, M, H, H, 0);
  gemmNT(xb, wkp, bk, nullptr, kf, nullptr, nullptr, M, H, H, 0);
  l2norm_kernel<<<M / 8, blk, 0, stream>>>(qf, qb, H);
  l2norm_kernel<<<M / 8, blk, 0, stream>>>(kf, kb, H);
  gemmNT(xb, wvp, bv, nullptr, vf, nullptr, nullptr, M, H, H, 0);
  gates_kernel<<<M / 8, blk, 0, stream>>>(x, wlr, blr, wf, bfp, wm, bm, lrv, sums, H);

  // --- Phase 2: memory forward on k (save z, h in bf16) ----------------------
  gemmNT(kb, mw1p, mb1, nullptr, nullptr, h0b, z0b, M, HE, H, 1);                 // z0,h0
  gemmNT(h0b, mw2p, mb2, kf, x1f, x1b, nullptr, M, H, HE, 0);                     // x1
  gemmNT(x1b, mw1p + W1SZ, mb1 + HE, nullptr, nullptr, h1b, z1b, M, HE, H, 1);   // z1,h1
  gemmNT(h1b, mw2p + W1SZ, mb2 + H, x1f, predf, nullptr, nullptr, M, H, HE, 0);  // pred

  // --- Phase 3: backward ------------------------------------------------------
  const float dscale = 2.f / ((float)H * (float)Cc);
  dpred_kernel<<<ew(MH), blk, 0, stream>>>(predf, vf, lrv, dpredf, dpb, H, dscale, MH);

  // depth 1
  gemmNT(dpb, mw2T + W1SZ, nullptr, nullptr, dhf, nullptr, nullptr, M, HE, H, 0);  // dh1
  dsilu_kernel<<<ew(MHE), blk, 0, stream>>>(dhf, z1b, dz1b, MHE);                  // dz1
  gemmTN(dpb, h1b, gw2 + W1SZ, M, H, HE);                                          // dW2[1]
  colsum_kernel<float><<<H, blk, 0, stream>>>(dpredf, gb2 + H, M, H);              // db2[1]
  gemmTN(dz1b, x1b, gw1 + W1SZ, M, HE, H);                                         // dW1[1]
  colsum_kernel<bf16><<<HE, blk, 0, stream>>>(dz1b, gb1 + HE, M, HE);              // db1[1]
  gemmNT(dz1b, mw1T + W1SZ, nullptr, dpredf, nullptr, dx1b, nullptr, M, H, HE, 0); // dx1

  // depth 0
  gemmNT(dx1b, mw2T, nullptr, nullptr, dhf, nullptr, nullptr, M, HE, H, 0);        // dh0
  dsilu_kernel<<<ew(MHE), blk, 0, stream>>>(dhf, z0b, dz0b, MHE);                  // dz0
  gemmTN(dx1b, h0b, gw2, M, H, HE);                                                // dW2[0]
  colsum_kernel<bf16><<<H, blk, 0, stream>>>(dx1b, gb2, M, H);                     // db2[0]
  gemmTN(dz0b, kb, gw1, M, HE, H);                                                 // dW1[0]
  colsum_kernel<bf16><<<HE, blk, 0, stream>>>(dz0b, gb1, M, HE);                   // db1[0]

  // --- Phase 4: scalar-gated update ------------------------------------------
  const float invM = 1.f / (float)M;
  update_kernel<<<ew(2 * W1SZ), blk, 0, stream>>>(mw1, gw1, mom_w1, sums, invM,
                                                  nullptr, nmw1p, 2 * W1SZ);
  update_kernel<<<ew(2 * (size_t)HE), blk, 0, stream>>>(mb1, gb1, mom_b1, sums, invM,
                                                        nmb1, nullptr, 2 * (size_t)HE);
  update_kernel<<<ew(2 * W1SZ), blk, 0, stream>>>(mw2, gw2, mom_w2, sums, invM,
                                                  nullptr, nmw2p, 2 * W1SZ);
  update_kernel<<<ew(2 * (size_t)H), blk, 0, stream>>>(mb2, gb2, mom_b2, sums, invM,
                                                       nmb2, nullptr, 2 * (size_t)H);

  // --- Phase 5: final forward on q with updated memory ------------------------
  gemmNT(qb, nmw1p, nmb1, nullptr, nullptr, h0b, nullptr, M, HE, H, 1);            // h0(q)
  gemmNT(h0b, nmw2p, nmb2, qf, dpredf, dpb, nullptr, M, H, HE, 0);                 // x1(q)
  gemmNT(dpb, nmw1p + W1SZ, nmb1 + HE, nullptr, nullptr, h1b, nullptr, M, HE, H, 1);
  gemmNT(h1b, nmw2p + W1SZ, nmb2 + H, dpredf, out, nullptr, nullptr, M, H, HE, 0); // output
}
